// OneModel_11098195493009
// MI455X (gfx1250) — matrix operand order repaired and checked by reading, not compiled, not run on hardware
//
#include <hip/hip_runtime.h>
#include <math.h>

typedef __attribute__((ext_vector_type(16))) _Float16 v16h;
typedef __attribute__((ext_vector_type(8)))  _Float16 v8h;
typedef __attribute__((ext_vector_type(8)))  float    v8f;

#define D_N1     900
#define D_NP1    912      // 57*16
#define D_NP2    928      // 29*32
#define D_MODEL  512
#define D_HEADS  8
#define D_HD     64
#define D_B      2
#define D_BH     16
#define D_MROWS  1800
#define D_MPAD   1824     // row padding for q/k/attn (b*900 + i, i<912)
#define D_MPF    1808     // 113*16, row padding for F/wp_out
#define D_MTILES 113
#define SCORE_SCALE 810000.0f   // n1*n2

// ---------------- WMMA helpers (CDNA5 v_wmma_f32_16x16x32_f16) ----------------

__device__ __forceinline__ v8f wmma_step(v16h a, v16h b, v8f c) {
  // (neg_a, A, neg_b, B, c_mod, C, reuse_a, reuse_b)
  return __builtin_amdgcn_wmma_f32_16x16x32_f16(false, a, false, b, (short)0, c,
                                                false, false);
}

// A-fragment (16x32 f16): lane holds row M=lane&15; halves 0..15 = K kb..kb+15
// with kb = (lane>>4)*16. Caller passes p already
// offset to (row, k0+kb): two contiguous 16B loads.
__device__ __forceinline__ v16h load_a16(const _Float16* p) {
  v8h lo = *(const v8h*)(p);
  v8h hi = *(const v8h*)(p + 8);
  return __builtin_shufflevector(lo, hi, 0,1,2,3,4,5,6,7,8,9,10,11,12,13,14,15);
}

__device__ __forceinline__ float wave_max(float x) {
#pragma unroll
  for (int o = 16; o; o >>= 1) x = fmaxf(x, __shfl_xor(x, o));
  return x;
}
__device__ __forceinline__ float wave_sum(float x) {
#pragma unroll
  for (int o = 16; o; o >>= 1) x += __shfl_xor(x, o);
  return x;
}

// ---------------- fp32 -> f16 hi/lo split (fp32 emulation inputs) ----------------

__global__ void cvt_split(const float* __restrict__ src, _Float16* __restrict__ hi,
                          _Float16* __restrict__ lo, int n) {
  int i = blockIdx.x * blockDim.x + threadIdx.x;
  if (i < n) {
    float x = src[i];
    _Float16 h = (_Float16)x;
    hi[i] = h;
    lo[i] = (_Float16)(x - (float)h);
  }
}

// ---------------- generic projection GEMM: C[M,512] = A[M,512] @ W[512,512]^T + b
// mode 0: write f32 (Wp). mode 1: per-64 l2norm (q/k), write hi/lo split.
// mode 2: no norm, write hi/lo split transposed vT[(b*8+h)*64+c][j]  (v).
// Grid: 113 blocks x 256 threads; wave w in block handles cols w*64..w*64+63 (= head w).

__global__ void proj_gemm(const _Float16* __restrict__ A_hi, const _Float16* __restrict__ A_lo,
                          const _Float16* __restrict__ W_hi, const _Float16* __restrict__ W_lo,
                          const float* __restrict__ bias, int mode, int Mvalid,
                          _Float16* __restrict__ out_hi, _Float16* __restrict__ out_lo,
                          float* __restrict__ out_f32) {
  const int lane = threadIdx.x & 31;
  const int w    = threadIdx.x >> 5;          // 0..7 col group / head
  const int row0 = blockIdx.x * 16;
  const int col0 = w * 64;
  const int m    = lane & 15;
  const int hiG  = lane >> 4;
  const int kbA  = hiG * 16;
  const int kbB  = hiG * 16;

  v8f zero = {0.f,0.f,0.f,0.f,0.f,0.f,0.f,0.f};
  v8f acc[4];
#pragma unroll
  for (int t = 0; t < 4; ++t) acc[t] = zero;

  const size_t aoff = (size_t)(row0 + m) * D_MODEL;
  for (int k0 = 0; k0 < D_MODEL; k0 += 32) {
    v16h ah = load_a16(A_hi + aoff + k0 + kbA);
    v16h al = load_a16(A_lo + aoff + k0 + kbA);
#pragma unroll
    for (int t = 0; t < 4; ++t) {
      size_t bo = (size_t)(col0 + t*16 + m) * D_MODEL + k0 + kbB;
      v16h bfh = *(const v16h*)(W_hi + bo);
      v16h bfl = *(const v16h*)(W_lo + bo);
      acc[t] = wmma_step(ah, bfh, acc[t]);
      acc[t] = wmma_step(al, bfh, acc[t]);
      acc[t] = wmma_step(ah, bfl, acc[t]);
    }
  }
  // bias (same column for all 8 rows of a VGPR)
#pragma unroll
  for (int t = 0; t < 4; ++t) {
    float bv_ = bias[col0 + t*16 + m];
#pragma unroll
    for (int jv = 0; jv < 8; ++jv) acc[t][jv] += bv_;
  }

  if (mode == 1) {                       // q/k: per-head l2 normalize, split store
#pragma unroll
    for (int jv = 0; jv < 8; ++jv) {
      float ss = 0.f;
#pragma unroll
      for (int t = 0; t < 4; ++t) ss += acc[t][jv] * acc[t][jv];
      ss += __shfl_xor(ss, 1); ss += __shfl_xor(ss, 2);
      ss += __shfl_xor(ss, 4); ss += __shfl_xor(ss, 8);
      float inv = 1.0f / fmaxf(sqrtf(ss), 1e-12f);
      int arow = row0 + jv + hiG * 8;
      if (arow < Mvalid) {
#pragma unroll
        for (int t = 0; t < 4; ++t) {
          float val = acc[t][jv] * inv;
          size_t idx = (size_t)arow * D_MODEL + col0 + t*16 + m;
          _Float16 h = (_Float16)val;
          out_hi[idx] = h; out_lo[idx] = (_Float16)(val - (float)h);
        }
      }
    }
  } else if (mode == 2) {                // v: store transposed vT[bh*64+c][j]
#pragma unroll
    for (int jv = 0; jv < 8; ++jv) {
      int arow = row0 + jv + hiG * 8;
      if (arow < Mvalid) {
        int b  = arow / D_N1;
        int jj = arow - b * D_N1;
#pragma unroll
        for (int t = 0; t < 4; ++t) {
          float val = acc[t][jv];
          size_t idx = ((size_t)(b * D_HEADS + w) * D_HD + t*16 + m) * D_NP2 + jj;
          _Float16 h = (_Float16)val;
          out_hi[idx] = h; out_lo[idx] = (_Float16)(val - (float)h);
        }
      }
    }
  } else {                               // Wp: raw f32 (padded rows allowed)
#pragma unroll
    for (int jv = 0; jv < 8; ++jv) {
      int arow = row0 + jv + hiG * 8;
#pragma unroll
      for (int t = 0; t < 4; ++t)
        out_f32[(size_t)arow * D_MODEL + col0 + t*16 + m] = acc[t][jv];
    }
  }
}

// ---------------- sim[bh][i][j] = q_i . k_j (padded region written 0) ----------------
// one wave = 16x32 tile; grid = 16*57*29 waves

__global__ void sim_gemm(const _Float16* __restrict__ q_hi, const _Float16* __restrict__ q_lo,
                         const _Float16* __restrict__ k_hi, const _Float16* __restrict__ k_lo,
                         float* __restrict__ simp, float beta) {
  int wid  = (blockIdx.x * blockDim.x + threadIdx.x) >> 5;
  int lane = threadIdx.x & 31;
  int jt = wid % 29;
  int it = (wid / 29) % 57;
  int bh = wid / (29 * 57);
  int b = bh >> 3, h = bh & 7;
  int i0 = it * 16, j0 = jt * 32;
  int m = lane & 15, hiG = lane >> 4;

  v8f zero = {0.f,0.f,0.f,0.f,0.f,0.f,0.f,0.f};
  v8f acc[2] = {zero, zero};
  size_t qoff = (size_t)(b * D_N1 + i0 + m) * D_MODEL + h * D_HD + hiG * 16;
#pragma unroll
  for (int k0 = 0; k0 < D_HD; k0 += 32) {
    v16h ah = load_a16(q_hi + qoff + k0);
    v16h al = load_a16(q_lo + qoff + k0);
#pragma unroll
    for (int s = 0; s < 2; ++s) {
      size_t ko = (size_t)(b * D_N1 + j0 + s*16 + m) * D_MODEL + h * D_HD + k0 + hiG * 16;
      v16h bfh = *(const v16h*)(k_hi + ko);
      v16h bfl = *(const v16h*)(k_lo + ko);
      acc[s] = wmma_step(ah, bfh, acc[s]);
      acc[s] = wmma_step(al, bfh, acc[s]);
      acc[s] = wmma_step(ah, bfl, acc[s]);
    }
  }
#pragma unroll
  for (int s = 0; s < 2; ++s)
#pragma unroll
    for (int jv = 0; jv < 8; ++jv) {
      int i  = i0 + jv + hiG * 8;
      int jj = j0 + s*16 + m;
      float val = (i < D_N1 && jj < D_N1) ? acc[s][jv] * beta : 0.0f;
      simp[((size_t)bh * D_NP1 + i) * D_NP2 + jj] = val;
    }
}

// ---------------- Sinkhorn ----------------
// u_new[bh,i] = eps*log_mu - eps*lse_j((sim-1+v_j)/eps)   (u cancels analytically)
// v_new[bh,j] = eps*log_nu - eps*lse_i((sim-1+u_i)/eps)   (uses pre-update u)

__global__ void init_state(float* u, float* unew, float* v, float* vnew, int* done) {
  int i = blockIdx.x * blockDim.x + threadIdx.x;
  if (i < D_BH * D_NP1) { u[i] = 0.f; unew[i] = 0.f; }
  if (i < D_BH * D_NP2) { v[i] = 0.f; vnew[i] = 0.f; }
  if (i == 0) *done = 0;
}

__global__ void sinkhorn_row(const float* __restrict__ simp, const float* __restrict__ vv,
                             float* __restrict__ u_new, const int* done,
                             float eps, float eps_log_mu) {
  if (*(volatile const int*)done) return;       // frozen once converged
  int wid  = (blockIdx.x * blockDim.x + threadIdx.x) >> 5;  // one wave per (bh,i)
  int lane = threadIdx.x & 31;
  int bh = wid / D_N1;
  int i  = wid - bh * D_N1;
  const float* row = simp + ((size_t)bh * D_NP1 + i) * D_NP2;
  const float* vp  = vv + bh * D_NP2;
  float inv_eps = 1.0f / eps;
  float mx = -1e30f;
  for (int j = lane; j < D_N1; j += 32) mx = fmaxf(mx, row[j] + vp[j]);
  mx = wave_max(mx);
  float s = 0.f;
  for (int j = lane; j < D_N1; j += 32) s += __expf((row[j] + vp[j] - mx) * inv_eps);
  s = wave_sum(s);
  if (lane == 0)
    u_new[bh * D_NP1 + i] = eps_log_mu - (mx - 1.0f) - eps * __logf(s);
}

__global__ void sinkhorn_col(const float* __restrict__ simp, const float* __restrict__ uu,
                             float* __restrict__ v_new, const int* done,
                             float eps, float eps_log_nu) {
  if (*(volatile const int*)done) return;
  int wid  = (blockIdx.x * blockDim.x + threadIdx.x) >> 5;  // one wave per (bh, 32 cols)
  int lane = threadIdx.x & 31;
  int bh = wid / 29;
  int jg = wid - bh * 29;
  int j  = jg * 32 + lane;                      // < 928; pad cols are zero-filled
  const float* base = simp + (size_t)bh * D_NP1 * D_NP2 + j;
  const float* up   = uu + bh * D_NP1;
  float inv_eps = 1.0f / eps;
  float mx = -1e30f;
  for (int i = 0; i < D_N1; ++i) mx = fmaxf(mx, base[(size_t)i * D_NP2] + up[i]);
  float s = 0.f;
  for (int i = 0; i < D_N1; ++i) s += __expf((base[(size_t)i * D_NP2] + up[i] - mx) * inv_eps);
  if (j < D_N1)
    v_new[bh * D_NP2 + j] = eps_log_nu - (mx - 1.0f) - eps * __logf(s);
}

__global__ void sinkhorn_update(float* __restrict__ u, const float* __restrict__ u_new,
                                float* __restrict__ v, const float* __restrict__ v_new,
                                int* done, float thresh) {
  if (*(volatile const int*)done) return;
  int tid = threadIdx.x;                        // single block, 1024 threads
  float acc = 0.f;
  for (int n = tid; n < D_BH * D_N1; n += 1024) {
    int bh = n / D_N1, i = n - bh * D_N1;
    int off = bh * D_NP1 + i;
    float un = u_new[off];
    acc += fabsf(un - u[off]);
    u[off] = un;
  }
  for (int n = tid; n < D_BH * D_N1; n += 1024) {
    int bh = n / D_N1, j = n - bh * D_N1;
    int off = bh * D_NP2 + j;
    v[off] = v_new[off];
  }
  __shared__ float sred[32];
  acc = wave_sum(acc);
  int wv = tid >> 5, lane = tid & 31;
  if (lane == 0) sred[wv] = acc;
  __syncthreads();
  if (wv == 0) {
    float x = sred[lane];
    x = wave_sum(x);
    if (lane == 0 && (x / (float)(D_BH * D_N1)) < thresh) *done = 1;
  }
}

// ---------------- score = n1*n2*sim*exp((sim-1+u+v)/eps); attn^T = vT @ score^T ----
// one wave per (bh, itile): out[c=0..63, i=i0..i0+15], K-loop over j (928/32 chunks)

__global__ void score_pv(const float* __restrict__ simp, const float* __restrict__ uu,
                         const float* __restrict__ vv,
                         const _Float16* __restrict__ vT_hi, const _Float16* __restrict__ vT_lo,
                         _Float16* __restrict__ attn_hi, _Float16* __restrict__ attn_lo,
                         float eps) {
  int wid  = (blockIdx.x * blockDim.x + threadIdx.x) >> 5;
  int lane = threadIdx.x & 31;
  int bh = wid / 57;
  int it = wid - bh * 57;
  int b = bh >> 3, h = bh & 7;
  int i0 = it * 16;
  int m = lane & 15, hiG = lane >> 4;
  int iB = i0 + m;                              // B-frag column (i), < 912
  float ui = uu[bh * D_NP1 + iB];
  float inv_eps = 1.0f / eps;

  v8f zero = {0.f,0.f,0.f,0.f,0.f,0.f,0.f,0.f};
  v8f acc[4] = {zero, zero, zero, zero};
  const float* srow = simp + ((size_t)bh * D_NP1 + iB) * D_NP2;
  const float* vp   = vv + bh * D_NP2;

  for (int j0 = 0; j0 < D_NP2; j0 += 32) {
    int jb = j0 + hiG * 16;                     // B-frag halves = contiguous j
    v16h sb_hi, sb_lo;
#pragma unroll
    for (int hh = 0; hh < 16; ++hh) {
      float sv = srow[jb + hh];                 // padded sim is 0 -> score 0
      float sc = SCORE_SCALE * sv * __expf((sv - 1.0f + ui + vp[jb + hh]) * inv_eps);
      _Float16 hx = (_Float16)sc;
      sb_hi[hh] = hx;
      sb_lo[hh] = (_Float16)(sc - (float)hx);
    }
#pragma unroll
    for (int t = 0; t < 4; ++t) {
      size_t ao = ((size_t)bh * D_HD + t*16 + m) * D_NP2 + j0 + hiG * 16;
      v16h ah = load_a16(vT_hi + ao);
      v16h al = load_a16(vT_lo + ao);
      acc[t] = wmma_step(ah, sb_hi, acc[t]);
      acc[t] = wmma_step(al, sb_hi, acc[t]);
      acc[t] = wmma_step(ah, sb_lo, acc[t]);
    }
  }
  int i = i0 + m;
  if (i < D_N1) {
    size_t arow = (size_t)b * D_N1 + i;
#pragma unroll
    for (int t = 0; t < 4; ++t)
#pragma unroll
      for (int jv = 0; jv < 8; ++jv) {
        int c = t*16 + jv + hiG * 8;
        float val = acc[t][jv];
        size_t idx = arow * D_MODEL + h * D_HD + c;
        _Float16 hx = (_Float16)val;
        attn_hi[idx] = hx; attn_lo[idx] = (_Float16)(val - (float)hx);
      }
  }
}

// ---------------- final row normalization (exact divide, no epsilon) ----------------

__global__ void row_normalize(const float* __restrict__ src, float* __restrict__ dst) {
  int wid  = (blockIdx.x * blockDim.x + threadIdx.x) >> 5;  // one wave per row, wid<1800
  int lane = threadIdx.x & 31;
  const float* p = src + (size_t)wid * D_MODEL;
  float ss = 0.f;
  for (int c = lane; c < D_MODEL; c += 32) { float x = p[c]; ss += x * x; }
  ss = wave_sum(ss);
  float inv = 1.0f / sqrtf(ss);
  for (int c = lane; c < D_MODEL; c += 32)
    dst[(size_t)wid * D_MODEL + c] = p[c] * inv;
}

// ---------------- launcher ----------------

extern "C" void kernel_launch(void* const* d_in, const int* in_sizes, int n_in,
                              void* d_out, int out_size, void* d_ws, size_t ws_size,
                              hipStream_t stream) {
  (void)in_sizes; (void)n_in; (void)out_size; (void)ws_size;
  const float* F_t = (const float*)d_in[0];
  const float* F_s = (const float*)d_in[1];
  const float* Wq  = (const float*)d_in[2];
  const float* bq  = (const float*)d_in[3];
  const float* Wk  = (const float*)d_in[4];
  const float* bk  = (const float*)d_in[5];
  const float* Wv  = (const float*)d_in[6];
  const float* bv  = (const float*)d_in[7];
  const float* Wp  = (const float*)d_in[8];
  const float* bp  = (const float*)d_in[9];
  float* out = (float*)d_out;

  char* base = (char*)d_ws;
  size_t off = 0;
  auto carve = [&](size_t bytes) -> void* {
    off = (off + 255) & ~(size_t)255;
    void* p = base + off;
    off += bytes;
    return p;
  };
  _Float16* Ft_hi = (_Float16*)carve((size_t)D_MPF * D_MODEL * 2);
  _Float16* Ft_lo = (_Float16*)carve((size_t)D_MPF * D_MODEL * 2);
  _Float16* Fs_hi = (_Float16*)carve((size_t)D_MPF * D_MODEL * 2);
  _Float16* Fs_lo = (_Float16*)carve((size_t)D_MPF * D_MODEL * 2);
  _Float16* Wq_hi = (_Float16*)carve((size_t)D_MODEL * D_MODEL * 2);
  _Float16* Wq_lo = (_Float16*)carve((size_t)D_MODEL * D_MODEL * 2);
  _Float16* Wk_hi = (_Float16*)carve((size_t)D_MODEL * D_MODEL * 2);
  _Float16* Wk_lo = (_Float16*)carve((size_t)D_MODEL * D_MODEL * 2);
  _Float16* Wv_hi = (_Float16*)carve((size_t)D_MODEL * D_MODEL * 2);
  _Float16* Wv_lo = (_Float16*)carve((size_t)D_MODEL * D_MODEL * 2);
  _Float16* Wp_hi = (_Float16*)carve((size_t)D_MODEL * D_MODEL * 2);
  _Float16* Wp_lo = (_Float16*)carve((size_t)D_MODEL * D_MODEL * 2);
  _Float16* q_hi  = (_Float16*)carve((size_t)D_MPAD * D_MODEL * 2);
  _Float16* q_lo  = (_Float16*)carve((size_t)D_MPAD * D_MODEL * 2);
  _Float16* k_hi  = (_Float16*)carve((size_t)D_MPAD * D_MODEL * 2);
  _Float16* k_lo  = (_Float16*)carve((size_t)D_MPAD * D_MODEL * 2);
  _Float16* vT_hi = (_Float16*)carve((size_t)D_BH * D_HD * D_NP2 * 2);
  _Float16* vT_lo = (_Float16*)carve((size_t)D_BH * D_HD * D_NP2 * 2);
  float*    simp  = (float*)carve((size_t)D_BH * D_NP1 * D_NP2 * 4);
  float*    d_u   = (float*)carve((size_t)D_BH * D_NP1 * 4);
  float*    d_un  = (float*)carve((size_t)D_BH * D_NP1 * 4);
  float*    d_v   = (float*)carve((size_t)D_BH * D_NP2 * 4);
  float*    d_vn  = (float*)carve((size_t)D_BH * D_NP2 * 4);
  _Float16* at_hi = (_Float16*)carve((size_t)D_MPAD * D_MODEL * 2);
  _Float16* at_lo = (_Float16*)carve((size_t)D_MPAD * D_MODEL * 2);
  float*    wp_o  = (float*)carve((size_t)D_MPF * D_MODEL * 4);
  int*      done  = (int*)carve(256);

  const float eps = 0.05f;
  const float elm = (float)(0.05 * log(1.0 / 900.0 + 1e-8));  // eps*log(mu+1e-8), nu identical
  const float beta = 1.0f;                                    // BETA_T

  // split inputs into f16 hi/lo (fp32-emulation operands)
  int nF = D_MROWS * D_MODEL, nW = D_MODEL * D_MODEL;
  cvt_split<<<(nF + 255) / 256, 256, 0, stream>>>(F_t, Ft_hi, Ft_lo, nF);
  cvt_split<<<(nF + 255) / 256, 256, 0, stream>>>(F_s, Fs_hi, Fs_lo, nF);
  cvt_split<<<(nW + 255) / 256, 256, 0, stream>>>(Wq, Wq_hi, Wq_lo, nW);
  cvt_split<<<(nW + 255) / 256, 256, 0, stream>>>(Wk, Wk_hi, Wk_lo, nW);
  cvt_split<<<(nW + 255) / 256, 256, 0, stream>>>(Wv, Wv_hi, Wv_lo, nW);
  cvt_split<<<(nW + 255) / 256, 256, 0, stream>>>(Wp, Wp_hi, Wp_lo, nW);

  // projections (q,k normalized per head; v stored transposed)
  proj_gemm<<<D_MTILES, 256, 0, stream>>>(Ft_hi, Ft_lo, Wq_hi, Wq_lo, bq, 1, D_MROWS, q_hi, q_lo, nullptr);
  proj_gemm<<<D_MTILES, 256, 0, stream>>>(Fs_hi, Fs_lo, Wk_hi, Wk_lo, bk, 1, D_MROWS, k_hi, k_lo, nullptr);
  proj_gemm<<<D_MTILES, 256, 0, stream>>>(Fs_hi, Fs_lo, Wv_hi, Wv_lo, bv, 2, D_MROWS, vT_hi, vT_lo, nullptr);

  // similarity matrix (L2-resident, padded with zeros)
  sim_gemm<<<(D_BH * 57 * 29) / 8, 256, 0, stream>>>(q_hi, q_lo, k_hi, k_lo, simp, beta);

  // sinkhorn (device-side early-exit via done flag; state frozen once converged)
  init_state<<<(D_BH * D_NP2 + 255) / 256, 256, 0, stream>>>(d_u, d_un, d_v, d_vn, done);
  for (int itn = 0; itn < 100; ++itn) {
    sinkhorn_row<<<(D_BH * D_N1) / 8, 256, 0, stream>>>(simp, d_v, d_un, done, eps, elm);
    sinkhorn_col<<<(D_BH * 29) / 8 + ((D_BH * 29) % 8 ? 1 : 0), 256, 0, stream>>>(simp, d_u, d_vn, done, eps, elm);
    sinkhorn_update<<<1, 1024, 0, stream>>>(d_u, d_un, d_v, d_vn, done, 1e-6f);
  }

  // attention output (score computed on the fly, split-f16 WMMA against vT)
  score_pv<<<(D_BH * 57) / 8 + ((D_BH * 57) % 8 ? 1 : 0), 256, 0, stream>>>(
      simp, d_u, d_v, vT_hi, vT_lo, at_hi, at_lo, eps);

  // output projection + final normalization
  proj_gemm<<<D_MTILES, 256, 0, stream>>>(at_hi, at_lo, Wp_hi, Wp_lo, bp, 0, D_MROWS, nullptr, nullptr, wp_o);
  row_normalize<<<D_MROWS / 8, 256, 0, stream>>>(wp_o, out);
}
